// DownSampleToken_43267500540729
// MI455X (gfx1250) — matrix operand order repaired and checked by reading, not compiled, not run on hardware
//
#include <hip/hip_runtime.h>

typedef __bf16 bf16_t;
typedef __attribute__((ext_vector_type(16))) __bf16 v16bf;
typedef __attribute__((ext_vector_type(8)))  __bf16 v8bf;
typedef __attribute__((ext_vector_type(8)))  float  v8f;

#define B_   16
#define C_   128
#define N_   2048
#define NK_  2052
#define NKP_ 2080
#define M_   1024
#define INVS 0.08838834764831845f  /* 1/sqrt(128) */

__device__ __forceinline__ v8f wmma_bf16(v16bf a, v16bf b, v8f c) {
  return __builtin_amdgcn_wmma_f32_16x16x32_bf16(false, a, false, b, (short)0, c, false, false);
}

__device__ __forceinline__ v16bf make_v16(v8bf lo, v8bf hi) {
  union { v16bf v; v8bf h[2]; } u; u.h[0] = lo; u.h[1] = hi; return u.v;
}

// A fragment (16x32 bf16): lane m = lane&15, half = lane>>4.
// elements e<8 -> k = half*8+e ; e>=8 -> k = 16+half*8+(e-8). rowptr points at (row m, col k0).
__device__ __forceinline__ v16bf load_a(const bf16_t* rowptr, int half) {
  const bf16_t* p = rowptr + half * 8;
  return make_v16(*(const v8bf*)p, *(const v8bf*)(p + 16));
}
// B fragment (32x16 bf16): lane n = lane&15, half = lane>>4; k in same order as A. p points at (row k0, col n0), ld = row stride.
__device__ __forceinline__ v16bf load_b(const bf16_t* p, int ld) {
  int l = threadIdx.x & 31, n = l & 15, h = l >> 4;
  v16bf v;
  #pragma unroll
  for (int s = 0; s < 16; ++s) {
    int k = (s < 8) ? (h * 8 + s) : (16 + h * 8 + (s - 8));
    v[s] = p[(size_t)k * ld + n];
  }
  return v;
}

// ---------------- init: transpose weights to bf16 (c,o)-major, zero V pad rows ----------------
__global__ void k_init(const float* __restrict__ Wq, const float* __restrict__ Wk,
                       const float* __restrict__ Wv,
                       bf16_t* __restrict__ wtq, bf16_t* __restrict__ wtk,
                       bf16_t* __restrict__ wtv, bf16_t* __restrict__ Vb) {
  int i = blockIdx.x * blockDim.x + threadIdx.x;
  if (i < 128 * 128) {
    int o = i >> 7, c = i & 127;
    wtq[c * 128 + o] = (bf16_t)Wq[i];
    wtk[c * 128 + o] = (bf16_t)Wk[i];
    wtv[c * 128 + o] = (bf16_t)Wv[i];
  }
  const int pad = B_ * (NKP_ - NK_) * C_;  // 16*28*128
  if (i < pad) {
    int per = (NKP_ - NK_) * C_;
    int b = i / per, rem = i % per;
    Vb[(size_t)b * NKP_ * C_ + (size_t)NK_ * C_ + rem] = (bf16_t)0.f;
  }
}

// ---------------- projection: Q (t,o), K (o,t), V (t,o) in bf16 ----------------
__global__ void k_proj(const float* __restrict__ x, const float* __restrict__ bins,
                       const bf16_t* __restrict__ wtq, const bf16_t* __restrict__ wtk,
                       const bf16_t* __restrict__ wtv,
                       bf16_t* __restrict__ Qb, bf16_t* __restrict__ Kb, bf16_t* __restrict__ Vb) {
  int b = blockIdx.x / 129;
  int tt = blockIdx.x % 129;
  int tbase = tt * 16;
  int tid = threadIdx.x;
  int lane = tid & 31, wave = tid >> 5, m16 = lane & 15, half = lane >> 4;

  __shared__ bf16_t sx[16][128];
  {  // stage x_tok^T tile (16 tokens x 128 channels) as bf16
    int c = tid;
    const float* xb = x + ((size_t)b * C_ + c) * N_;
    const float* bp = bins + (size_t)c * 4;
    #pragma unroll
    for (int tl = 0; tl < 16; ++tl) {
      int t = tbase + tl;
      float v = 0.f;
      if (t < N_) v = xb[t];
      else if (t < NK_) v = bp[t - N_];
      sx[tl][c] = (bf16_t)v;
    }
  }
  __syncthreads();

  v16bf a[4];
  #pragma unroll
  for (int kk = 0; kk < 4; ++kk) a[kk] = load_a(&sx[m16][kk * 32], half);

  bf16_t* Qp = Qb + (size_t)b * N_ * C_;
  bf16_t* Kp = Kb + (size_t)b * C_ * NKP_;
  bf16_t* Vp = Vb + (size_t)b * NKP_ * C_;

  for (int mat = 0; mat < 3; ++mat) {
    if (mat == 0 && tbase >= N_) continue;
    const bf16_t* W = (mat == 0) ? wtq : ((mat == 1) ? wtk : wtv);
    #pragma unroll
    for (int oi = 0; oi < 2; ++oi) {
      int ot = wave * 2 + oi;
      v8f acc = {};
      #pragma unroll
      for (int kk = 0; kk < 4; ++kk)
        acc = wmma_bf16(a[kk], load_b(W + (size_t)(kk * 32) * 128 + ot * 16, 128), acc);
      int o = ot * 16 + m16;
      #pragma unroll
      for (int r = 0; r < 8; ++r) {
        int t = tbase + r + 8 * half;
        if (mat == 0)      { if (t < N_)  Qp[(size_t)t * C_ + o]   = (bf16_t)acc[r]; }
        else if (mat == 1) { if (t < NK_) Kp[(size_t)o * NKP_ + t] = (bf16_t)acc[r]; }
        else               { if (t < NK_) Vp[(size_t)t * C_ + o]   = (bf16_t)acc[r]; }
      }
    }
  }
}

// ---------------- attention stats: rowmax/rowsum (flash), key partial scores, bin energies ----------------
__global__ void k_stats(const bf16_t* __restrict__ Qb, const bf16_t* __restrict__ Kb,
                        float* __restrict__ scorep, float* __restrict__ ebins,
                        float* __restrict__ rowm, float* __restrict__ rowl) {
  int b = blockIdx.x >> 7, qt = blockIdx.x & 127;
  int tid = threadIdx.x, lane = tid & 31, wave = tid >> 5, m16 = lane & 15, half = lane >> 4;
  const bf16_t* Qp = Qb + (size_t)b * N_ * C_;
  const bf16_t* Kp = Kb + (size_t)b * C_ * NKP_;

  v16bf a[4];
  #pragma unroll
  for (int kk = 0; kk < 4; ++kk)
    a[kk] = load_a(Qp + (size_t)(qt * 16 + m16) * C_ + kk * 32, half);

  float mrun[8], lrun[8];
  #pragma unroll
  for (int r = 0; r < 8; ++r) { mrun[r] = -__builtin_inff(); lrun[r] = 0.f; }

  // pass A: online max/sumexp over all 2052 real keys (129 key tiles)
  for (int t = wave; t < 129; t += 4) {
    v8f c = {};
    #pragma unroll
    for (int kk = 0; kk < 4; ++kk)
      c = wmma_bf16(a[kk], load_b(Kp + (size_t)(kk * 32) * NKP_ + t * 16, NKP_), c);
    int key = t * 16 + m16;
    if (key < NK_) {
      #pragma unroll
      for (int r = 0; r < 8; ++r) {
        float e = c[r] * INVS;
        if (key >= N_)
          ebins[((size_t)b * N_ + qt * 16 + r + 8 * half) * 4 + (key - N_)] = e;
        float mn = fmaxf(mrun[r], e);
        lrun[r] = lrun[r] * __expf(mrun[r] - mn) + __expf(e - mn);
        mrun[r] = mn;
      }
    }
  }
  // merge the 16 lanes of each half
  #pragma unroll
  for (int off = 1; off < 16; off <<= 1) {
    #pragma unroll
    for (int r = 0; r < 8; ++r) {
      float mo = __shfl_xor(mrun[r], off, 32);
      float lo = __shfl_xor(lrun[r], off, 32);
      float mn = fmaxf(mrun[r], mo);
      lrun[r] = lrun[r] * __expf(mrun[r] - mn) + lo * __expf(mo - mn);
      mrun[r] = mn;
    }
  }
  __shared__ float smx[4][16], sls[4][16], rowM[16], rowL[16];
  if (m16 == 0) {
    #pragma unroll
    for (int r = 0; r < 8; ++r) { smx[wave][r + 8 * half] = mrun[r]; sls[wave][r + 8 * half] = lrun[r]; }
  }
  __syncthreads();
  if (tid < 16) {
    float M = -__builtin_inff(), L = 0.f;
    #pragma unroll
    for (int w = 0; w < 4; ++w) {
      float mw = smx[w][tid], lw = sls[w][tid];
      float mn = fmaxf(M, mw);
      L = L * __expf(M - mn) + lw * __expf(mw - mn);
      M = mn;
    }
    rowM[tid] = M; rowL[tid] = L;
    rowm[(size_t)b * N_ + qt * 16 + tid] = M;
    rowl[(size_t)b * N_ + qt * 16 + tid] = L;
  }
  __syncthreads();
  float Mv[8], Lv[8];
  #pragma unroll
  for (int r = 0; r < 8; ++r) { Mv[r] = rowM[r + 8 * half]; Lv[r] = rowL[r + 8 * half]; }

  // pass B: recompute energies, emit per-(qtile,key) partial column sums (keys < 2048 only)
  for (int t = wave; t < 128; t += 4) {
    v8f c = {};
    #pragma unroll
    for (int kk = 0; kk < 4; ++kk)
      c = wmma_bf16(a[kk], load_b(Kp + (size_t)(kk * 32) * NKP_ + t * 16, NKP_), c);
    float cs = 0.f;
    #pragma unroll
    for (int r = 0; r < 8; ++r) cs += __expf(c[r] * INVS - Mv[r]) / Lv[r];
    cs += __shfl_xor(cs, 16, 32);
    if (lane < 16) scorep[((size_t)b * 128 + qt) * N_ + t * 16 + m16] = cs;
  }
}

// ---------------- deterministic score reduction ----------------
__global__ void k_reduce(const float* __restrict__ scorep, float* __restrict__ score) {
  int i = blockIdx.x * blockDim.x + threadIdx.x;
  if (i >= B_ * N_) return;
  int b = i / N_, n = i % N_;
  const float* p = scorep + (size_t)b * 128 * N_ + n;
  float s = 0.f;
  for (int q = 0; q < 128; ++q) s += p[(size_t)q * N_];
  score[i] = s;
}

// ---------------- bin stats, water-fill, stable ranking, prefix-scan selection ----------------
__global__ void k_select(const float* __restrict__ score, const float* __restrict__ ebins,
                         int* __restrict__ idxbuf, int* __restrict__ outIdx) {
  int b = blockIdx.x, tid = threadIdx.x;
  __shared__ float ssc[N_];
  __shared__ unsigned char sbin[N_];
  __shared__ float pw[256][4];
  __shared__ int   pc[256][4];
  __shared__ int   kb[4];
  __shared__ int   scnt[256];

  float lw[4] = {0, 0, 0, 0}; int lc[4] = {0, 0, 0, 0};
  for (int n = tid; n < N_; n += 256) {
    float s = score[(size_t)b * N_ + n];
    int bi = (s >= 1.2f) ? 0 : (s >= 1.0f ? 1 : (s >= 0.8f ? 2 : 3));
    ssc[n] = s; sbin[n] = (unsigned char)bi;
    lw[bi] += ebins[((size_t)b * N_ + n) * 4 + bi];
    lc[bi] += 1;
  }
  #pragma unroll
  for (int j = 0; j < 4; ++j) { pw[tid][j] = lw[j]; pc[tid][j] = lc[j]; }
  #pragma unroll
  for (int j = 0; j < 4; ++j) { int p = tid * 4 + j; idxbuf[(size_t)b * M_ + p] = 0; outIdx[(size_t)b * M_ + p] = 0; }
  __syncthreads();
  if (tid < 4) {
    float w = 0.f; int c = 0;
    for (int t = 0; t < 256; ++t) { w += pw[t][tid]; c += pc[t][tid]; }
    pw[0][tid] = w; pc[0][tid] = c;
  }
  __syncthreads();
  if (tid == 0) {
    float bw[4]; int mp[4], k0[4];
    float tot = 0.f;
    for (int j = 0; j < 4; ++j) {
      mp[j] = pc[0][j];
      float wp = pw[0][j] / ((float)mp[j] + 1e-8f);
      bw[j] = wp > 0.f ? wp : 0.f;
      tot += bw[j];
    }
    tot = fmaxf(tot, 1e-8f);
    int sk = 0;
    for (int j = 0; j < 4; ++j) {
      k0[j] = (int)floorf(bw[j] / tot * (float)M_);
      if (k0[j] > mp[j]) k0[j] = mp[j];
      sk += k0[j];
    }
    int def = M_ - sk, cum = 0;
    for (int j = 0; j < 4; ++j) {
      int cap = mp[j] - k0[j];
      int ex = def - cum; if (ex < 0) ex = 0; if (ex > cap) ex = cap;
      kb[j] = k0[j] + ex; cum += cap;
    }
  }
  __syncthreads();
  // stable rank within bin (score desc, index asc on ties) — matches argsort semantics
  int cnt = 0; unsigned int flags = 0;
  for (int i = 0; i < 8; ++i) {
    int n = tid * 8 + i;
    float s = ssc[n]; int bi = sbin[n];
    int rank = 0;
    for (int j = 0; j < N_; ++j) {
      if (sbin[j] == bi) {
        float sj = ssc[j];
        if (sj > s || (sj == s && j < n)) ++rank;
      }
    }
    if (rank < kb[bi]) { flags |= (1u << i); ++cnt; }
  }
  scnt[tid] = cnt;
  __syncthreads();
  if (tid == 0) {
    int run = 0;
    for (int t = 0; t < 256; ++t) { int c2 = scnt[t]; scnt[t] = run; run += c2; }
  }
  __syncthreads();
  int pos = scnt[tid];
  for (int i = 0; i < 8; ++i) {
    if (flags & (1u << i)) {
      if (pos < M_) {
        idxbuf[(size_t)b * M_ + pos] = tid * 8 + i;
        outIdx[(size_t)b * M_ + pos] = tid * 8 + i;
      }
      ++pos;
    }
  }
}

// ---------------- gathered attention rows + attn·V, x_ds transposed store ----------------
__global__ void k_gather(const bf16_t* __restrict__ Qb, const bf16_t* __restrict__ Kb,
                         const bf16_t* __restrict__ Vb,
                         const float* __restrict__ rowm, const float* __restrict__ rowl,
                         const int* __restrict__ idxbuf, float* __restrict__ out) {
  int b = blockIdx.x >> 6, tile = blockIdx.x & 63;
  int tid = threadIdx.x, lane = tid & 31, wave = tid >> 5, m16 = lane & 15, half = lane >> 4;
  const bf16_t* Qp = Qb + (size_t)b * N_ * C_;
  const bf16_t* Kp = Kb + (size_t)b * C_ * NKP_;
  const bf16_t* Vp = Vb + (size_t)b * NKP_ * C_;

  __shared__ int   sidx[16];
  __shared__ float sM[16], sL[16];
  if (tid < 16) {
    int q = idxbuf[(size_t)b * M_ + tile * 16 + tid];
    if (q < 0) q = 0; if (q >= N_) q = N_ - 1;
    sidx[tid] = q;
    sM[tid] = rowm[(size_t)b * N_ + q];
    sL[tid] = rowl[(size_t)b * N_ + q];
  }
  __syncthreads();

  int qrow = sidx[m16];
  v16bf a[4];
  #pragma unroll
  for (int kk = 0; kk < 4; ++kk)
    a[kk] = load_a(Qp + (size_t)qrow * C_ + kk * 32, half);
  float Mv[8], Lv[8];
  #pragma unroll
  for (int r = 0; r < 8; ++r) { Mv[r] = sM[r + 8 * half]; Lv[r] = sL[r + 8 * half]; }

  const int LW = 1072;                // padded row stride (bf16 elems)
  __shared__ bf16_t sAttn[16 * LW];   // 33.5 KB

  v8f acc0 = {}, acc1 = {};
  int dt0 = wave * 2, dt1 = wave * 2 + 1;

  #pragma unroll
  for (int chunk = 0; chunk < 2; ++chunk) {
    int t0 = chunk ? 64 : 0;
    int t1 = chunk ? 130 : 64;
    int kbase = chunk ? 1024 : 0;
    int kIters = chunk ? 33 : 32;
    for (int t = t0 + wave; t < t1; t += 4) {
      v8f c = {};
      #pragma unroll
      for (int kk = 0; kk < 4; ++kk)
        c = wmma_bf16(a[kk], load_b(Kp + (size_t)(kk * 32) * NKP_ + t * 16, NKP_), c);
      int key = t * 16 + m16;
      int loc = key - kbase;
      #pragma unroll
      for (int r = 0; r < 8; ++r) {
        float att = 0.f;
        if (key < NK_) att = __expf(c[r] * INVS - Mv[r]) / Lv[r];
        sAttn[(r + 8 * half) * LW + loc] = (bf16_t)att;
      }
    }
    __syncthreads();
    const bf16_t* arow = &sAttn[m16 * LW];
    for (int kk = 0; kk < kIters; ++kk) {
      v16bf af = load_a(arow + kk * 32, half);
      size_t vr = (size_t)(kbase + kk * 32) * C_;
      acc0 = wmma_bf16(af, load_b(Vp + vr + dt0 * 16, C_), acc0);
      acc1 = wmma_bf16(af, load_b(Vp + vr + dt1 * 16, C_), acc1);
    }
    __syncthreads();
  }
  int d0 = dt0 * 16 + m16, d1 = dt1 * 16 + m16;
  #pragma unroll
  for (int r = 0; r < 8; ++r) {
    int mpos = tile * 16 + r + 8 * half;
    out[((size_t)b * C_ + d0) * M_ + mpos] = acc0[r];
    out[((size_t)b * C_ + d1) * M_ + mpos] = acc1[r];
  }
}

extern "C" void kernel_launch(void* const* d_in, const int* in_sizes, int n_in,
                              void* d_out, int out_size, void* d_ws, size_t ws_size,
                              hipStream_t stream) {
  (void)in_sizes; (void)n_in; (void)out_size; (void)ws_size;
  const float* x    = (const float*)d_in[0];
  const float* Wq   = (const float*)d_in[1];
  const float* Wk   = (const float*)d_in[2];
  const float* Wv   = (const float*)d_in[3];
  const float* bins = (const float*)d_in[4];

  char* ws = (char*)d_ws;
  size_t off = 0;
  auto alloc = [&](size_t bytes) { size_t o = off; off += (bytes + 255) & ~(size_t)255; return o; };
  size_t oWtq = alloc(128 * 128 * 2);
  size_t oWtk = alloc(128 * 128 * 2);
  size_t oWtv = alloc(128 * 128 * 2);
  size_t oQ   = alloc((size_t)B_ * N_ * C_ * 2);
  size_t oK   = alloc((size_t)B_ * C_ * NKP_ * 2);
  size_t oV   = alloc((size_t)B_ * NKP_ * C_ * 2);
  size_t oSP  = alloc((size_t)B_ * 128 * N_ * 4);
  size_t oSC  = alloc((size_t)B_ * N_ * 4);
  size_t oEB  = alloc((size_t)B_ * N_ * 4 * 4);
  size_t oRM  = alloc((size_t)B_ * N_ * 4);
  size_t oRL  = alloc((size_t)B_ * N_ * 4);
  size_t oIDX = alloc((size_t)B_ * M_ * 4);

  bf16_t* wtq = (bf16_t*)(ws + oWtq);
  bf16_t* wtk = (bf16_t*)(ws + oWtk);
  bf16_t* wtv = (bf16_t*)(ws + oWtv);
  bf16_t* Qb  = (bf16_t*)(ws + oQ);
  bf16_t* Kb  = (bf16_t*)(ws + oK);
  bf16_t* Vb  = (bf16_t*)(ws + oV);
  float*  scorep = (float*)(ws + oSP);
  float*  score  = (float*)(ws + oSC);
  float*  ebins  = (float*)(ws + oEB);
  float*  rowm   = (float*)(ws + oRM);
  float*  rowl   = (float*)(ws + oRL);
  int*    idx    = (int*)(ws + oIDX);

  float* xds   = (float*)d_out;
  int*   oIdxO = (int*)d_out + (size_t)B_ * C_ * M_;

  k_init  <<<(B_ * (NKP_ - NK_) * C_ + 255) / 256, 256, 0, stream>>>(Wq, Wk, Wv, wtq, wtk, wtv, Vb);
  k_proj  <<<B_ * 129, 128, 0, stream>>>(x, bins, wtq, wtk, wtv, Qb, Kb, Vb);
  k_stats <<<B_ * 128, 128, 0, stream>>>(Qb, Kb, scorep, ebins, rowm, rowl);
  k_reduce<<<(B_ * N_ + 255) / 256, 256, 0, stream>>>(scorep, score);
  k_select<<<B_, 256, 0, stream>>>(score, ebins, idx, oIdxO);
  k_gather<<<B_ * 64, 128, 0, stream>>>(Qb, Kb, Vb, rowm, rowl, idx, xds);
}
